// MoEGate_83700322664573
// MI455X (gfx1250) — matrix operand order repaired and checked by reading, not compiled, not run on hardware
//
#include <hip/hip_runtime.h>

typedef __attribute__((ext_vector_type(16))) __bf16        v16bf;
typedef __attribute__((ext_vector_type(8)))  float         v8f;
typedef __attribute__((ext_vector_type(4)))  unsigned int  u32x4;
typedef __attribute__((ext_vector_type(4)))  float         f32x4;

#define S_TOK 16384
#define HDIM  7168
#define NEXP  256
#define BM    32
#define BK    64

// Pack two fp32 into a bf16x2 dword (RNE). Prefer the hardware cvt op.
__device__ __forceinline__ unsigned pk_bf16f(float a, float b) {
#if defined(__has_builtin) && __has_builtin(__builtin_amdgcn_cvt_pk_bf16_f32)
    typedef __attribute__((ext_vector_type(2))) __bf16 v2bf;
    v2bf r = __builtin_amdgcn_cvt_pk_bf16_f32(a, b);
    return __builtin_bit_cast(unsigned, r);
#else
    unsigned lo = __builtin_bit_cast(unsigned, a);
    unsigned hi = __builtin_bit_cast(unsigned, b);
    unsigned rl = (lo + 0x7FFFu + ((lo >> 16) & 1u)) >> 16;
    unsigned rh = (hi + 0x7FFFu + ((hi >> 16) & 1u)) & 0xFFFF0000u;
    return rl | rh;
#endif
}

// --- Kernel 1: convert router weight fp32 -> bf16 (row-major, packed) ---
__global__ __launch_bounds__(256) void wconv_kernel(const float* __restrict__ w,
                                                    unsigned* __restrict__ wbf) {
    const unsigned i = blockIdx.x * 256u + threadIdx.x;   // 8 fp32 elements per thread
    const f32x4* src = (const f32x4*)w + (size_t)i * 2;
    f32x4 f0 = src[0];
    f32x4 f1 = src[1];
    u32x4 o;
    o.x = pk_bf16f(f0.x, f0.y);
    o.y = pk_bf16f(f0.z, f0.w);
    o.z = pk_bf16f(f1.x, f1.y);
    o.w = pk_bf16f(f1.z, f1.w);
    ((u32x4*)wbf)[i] = o;
}

// --- Kernel 2: fused gating GEMM (bf16 WMMA) + sigmoid + grouped top-k ---
__global__ __launch_bounds__(256) void gate_kernel(const float* __restrict__ hs,
                                                   const unsigned short* __restrict__ wbf,
                                                   const float* __restrict__ bias,
                                                   float* __restrict__ out_w,
                                                   int* __restrict__ out_i) {
    __shared__ alignas(16) unsigned As[BM * (BK / 2)];   // 32 tokens x 64 k, bf16 packed (4KB)
    __shared__ float bias_s[NEXP];
    __shared__ float scores[BM][NEXP + 1];               // +1 pad: conflict-free column scans

    const int tid  = threadIdx.x;
    const int wave = tid >> 5;
    const int lane = tid & 31;
    const int half = lane >> 4;
    const int l16  = lane & 15;
    const int m0   = blockIdx.x * BM;

    if (tid < NEXP) bias_s[tid] = bias[tid];

    v8f c00 = {}, c01 = {}, c10 = {}, c11 = {};

    // cooperative A staging: thread loads 8 fp32 of row (tid>>3), cols (tid&7)*8
    const int arow = tid >> 3;
    const int acol = (tid & 7) << 3;
    const float* aptr = hs + (size_t)(m0 + arow) * HDIM + acol;

    // per-wave B: experts [wave*32, wave*32+32)
    const int e0 = wave << 5;
    const unsigned short* bbase0 = wbf + (size_t)(e0 + l16) * HDIM + half * 16;
    const unsigned short* bbase1 = wbf + (size_t)(e0 + 16 + l16) * HDIM + half * 16;

    union ABU { v16bf v; u32x4 u[2]; };

    for (int k = 0; k < HDIM; k += BK) {
        __syncthreads();
        {   // fp32 -> bf16 convert into LDS (one ds_store_b128 per thread)
            const f32x4 f0 = *(const f32x4*)(aptr + k);
            const f32x4 f1 = *(const f32x4*)(aptr + k + 4);
            u32x4 o;
            o.x = pk_bf16f(f0.x, f0.y);
            o.y = pk_bf16f(f0.z, f0.w);
            o.z = pk_bf16f(f1.x, f1.y);
            o.w = pk_bf16f(f1.z, f1.w);
            ((u32x4*)As)[arow * 8 + (tid & 7)] = o;
        }
        __syncthreads();

        const u32x4* Ab = (const u32x4*)As;
        #pragma unroll
        for (int s = 0; s < 2; ++s) {                    // two K=32 WMMA steps per stage
            // A fragments: half 0 holds K 0-15, half 1 holds K 16-31
            ABU a0, a1, b0, b1;
            a0.u[0] = Ab[l16 * 8        + s * 4 + 2 * half];
            a0.u[1] = Ab[l16 * 8        + s * 4 + 2 * half + 1];
            a1.u[0] = Ab[(l16 + 16) * 8 + s * 4 + 2 * half];
            a1.u[1] = Ab[(l16 + 16) * 8 + s * 4 + 2 * half + 1];

            // B fragments: half 0 holds K 0-15, half 1 holds K 16-31 (contiguous bf16)
            const u32x4* bp0 = (const u32x4*)(bbase0 + k + s * 32);
            const u32x4* bp1 = (const u32x4*)(bbase1 + k + s * 32);
            b0.u[0] = bp0[0]; b0.u[1] = bp0[1];
            b1.u[0] = bp1[0]; b1.u[1] = bp1[1];

            c00 = __builtin_amdgcn_wmma_f32_16x16x32_bf16(false, a0.v, false, b0.v, (short)0, c00, false, false);
            c01 = __builtin_amdgcn_wmma_f32_16x16x32_bf16(false, a0.v, false, b1.v, (short)0, c01, false, false);
            c10 = __builtin_amdgcn_wmma_f32_16x16x32_bf16(false, a1.v, false, b0.v, (short)0, c10, false, false);
            c11 = __builtin_amdgcn_wmma_f32_16x16x32_bf16(false, a1.v, false, b1.v, (short)0, c11, false, false);
        }
    }

    // C layout: VGPR j -> M = j + 8*half, N = l16
    #pragma unroll
    for (int j = 0; j < 8; ++j) {
        const int r0 = j + half * 8;
        scores[r0][e0 + l16]           = c00[j];
        scores[r0][e0 + 16 + l16]      = c01[j];
        scores[r0 + 16][e0 + l16]      = c10[j];
        scores[r0 + 16][e0 + 16 + l16] = c11[j];
    }
    __syncthreads();

    // ---- phase 2: per-token grouped top-k (one token per thread) ----
    if (tid < BM) {
        float* row = scores[tid];
        for (int e = 0; e < NEXP; ++e)
            row[e] = 1.0f / (1.0f + __expf(-row[e]));          // unbiased sigmoid scores

        float gs[8];                                            // group = sum of top-2 biased
        #pragma unroll
        for (int g = 0; g < 8; ++g) {
            float m1 = -1e30f, m2 = -1e30f;
            for (int i = 0; i < 32; ++i) {
                const int e = (g << 5) + i;
                const float v = row[e] + bias_s[e];
                if (v > m1)      { m2 = m1; m1 = v; }
                else if (v > m2) { m2 = v; }
            }
            gs[g] = m1 + m2;
        }

        unsigned gmask = 0;                                     // top-4 groups
        #pragma unroll
        for (int t = 0; t < 4; ++t) {
            float best = -1e30f; int bg = 0;
            #pragma unroll
            for (int g = 0; g < 8; ++g)
                if (!((gmask >> g) & 1u) && gs[g] > best) { best = gs[g]; bg = g; }
            gmask |= 1u << bg;
        }

        float wv[8]; int wi[8]; float wsum = 0.0f;              // masked top-8 experts
        #pragma unroll
        for (int t = 0; t < 8; ++t) {
            float best = -1e30f; int bi = 0;
            for (int g = 0; g < 8; ++g) {
                if (!((gmask >> g) & 1u)) continue;
                for (int i = 0; i < 32; ++i) {
                    const int e = (g << 5) + i;
                    const float v = row[e] + bias_s[e];
                    if (v > best) { best = v; bi = e; }
                }
            }
            wv[t] = row[bi]; wi[t] = bi;
            row[bi] = -1e30f;                                   // exclude from next rounds
            wsum += wv[t];
        }

        const float scale = 2.5f / (wsum + 1e-20f);
        const int tok = m0 + tid;
        #pragma unroll
        for (int t = 0; t < 8; ++t) {
            out_w[tok * 8 + t] = wv[t] * scale;
            out_i[tok * 8 + t] = wi[t];
        }
    }
}

extern "C" void kernel_launch(void* const* d_in, const int* in_sizes, int n_in,
                              void* d_out, int out_size, void* d_ws, size_t ws_size,
                              hipStream_t stream) {
    const float* hs   = (const float*)d_in[0];   // [16384, 7168]
    const float* w    = (const float*)d_in[1];   // [256, 7168]
    const float* bias = (const float*)d_in[2];   // [256]

    unsigned short* wbf = (unsigned short*)d_ws; // 256*7168 bf16 = 3.67 MB scratch
    float* out_w = (float*)d_out;                // [16384, 8] weights
    int*   out_i = (int*)d_out + S_TOK * 8;      // [16384, 8] indices (concatenated)

    // 256*7168 = 1,835,008 elems; 8 per thread -> 896 blocks of 256
    wconv_kernel<<<896, 256, 0, stream>>>(w, (unsigned*)wbf);
    gate_kernel<<<S_TOK / BM, 256, 0, stream>>>(hs, wbf, bias, out_w, out_i);
}
